// ClauseGCN_43482248905086
// MI455X (gfx1250) — matrix operand order repaired and checked by reading, not compiled, not run on hardware
//
#include <hip/hip_runtime.h>

// ---------------------------------------------------------------------------
// ClauseGCN forward for MI455X (gfx1250, wave32)
//   embed -> W_in  (scalar f32: K=15, too small for WMMA, keeps full precision)
//   3x [ SpMM (f32 atomics, gathers hit 192MB L2) ->
//        128x128 GEMM via v_wmma_f32_16x16x32_bf16 (bf16 in, f32 accum,
//        4 row-tiles per wave with B fragments register-resident) ->
//        GraphNorm (two atomic segment reductions) -> ReLU ]
// ---------------------------------------------------------------------------

#define HID    128
#define GNUM   256
#define LAYERS 3
#define EPSV   1e-5f
#define MT     4            // row tiles per wave in the WMMA GEMM

typedef __attribute__((ext_vector_type(16))) __bf16 v16bf;
typedef __attribute__((ext_vector_type(8)))  float  v8f;

__device__ __forceinline__ unsigned short f32_to_bf16_rne(float f) {
  unsigned u = __float_as_uint(f);
  unsigned r = u + 0x7FFFu + ((u >> 16) & 1u);   // round-to-nearest-even
  return (unsigned short)(r >> 16);
}

// ---------------- zero ----------------
__global__ void zero_k(float* p, long long n) {
  long long i = (long long)blockIdx.x * blockDim.x + threadIdx.x;
  if (i < n) p[i] = 0.0f;
}

// ---------------- embed + input projection ----------------
// feats = [onehot(type,6), log1p(arity), sin/cos interleaved over 4 freqs]
// DIV_TERM = 10^{-0..-3} exactly.
#define NPB 32
__global__ void embed_k(const float* __restrict__ x, const float* __restrict__ W_in,
                        const float* __restrict__ b_in, float* __restrict__ H, int N_) {
  __shared__ float f[NPB][16];
  int t = threadIdx.x;                 // 0..127
  int base = blockIdx.x * NPB;
  if (t < NPB) {
    int n = base + t;
    if (n < N_) {
      float ty = x[n * 3 + 0], ar = x[n * 3 + 1], ap = x[n * 3 + 2];
      int it = (int)ty; it = min(max(it, 0), 5);
      #pragma unroll
      for (int k = 0; k < 6; ++k) f[t][k] = (k == it) ? 1.0f : 0.0f;
      f[t][6] = log1pf(ar);
      const float div[4] = {1.0f, 0.1f, 0.01f, 0.001f};
      #pragma unroll
      for (int j = 0; j < 4; ++j) {
        float s = ap * div[j];
        f[t][7 + 2 * j] = sinf(s);
        f[t][8 + 2 * j] = cosf(s);
      }
    }
  }
  __syncthreads();
  int j = t;                           // output feature 0..127
  float w[15];
  #pragma unroll
  for (int k = 0; k < 15; ++k) w[k] = W_in[k * HID + j];
  float bj = b_in[j];
  for (int i = 0; i < NPB; ++i) {
    int n = base + i;
    if (n >= N_) break;
    float acc = bj;
    #pragma unroll
    for (int k = 0; k < 15; ++k) acc = fmaf(f[i][k], w[k], acc);
    H[(size_t)n * HID + j] = acc;      // coalesced across j
  }
}

// ---------------- SpMM: agg[row] += val * h[col] ----------------
// 32 lanes per edge, float4 per lane -> fully coalesced 512B row gather.
__global__ void spmm_k(const int* __restrict__ er, const int* __restrict__ ec,
                       const float* __restrict__ ev, const float* __restrict__ h,
                       float* __restrict__ agg, int E_) {
  long long gid = (long long)blockIdx.x * blockDim.x + threadIdx.x;
  int e = (int)(gid >> 5);
  if (e >= E_) return;
  int seg = (int)(gid & 31);
  int r = er[e], c = ec[e];
  float v = ev[e];
  float4 hx = ((const float4*)(h + (size_t)c * HID))[seg];
  float* op = agg + (size_t)r * HID + seg * 4;
  unsafeAtomicAdd(op + 0, v * hx.x);
  unsafeAtomicAdd(op + 1, v * hx.y);
  unsafeAtomicAdd(op + 2, v * hx.z);
  unsafeAtomicAdd(op + 3, v * hx.w);
}

// ---------------- f32 -> bf16 (packed pairs) ----------------
__global__ void cvt_bf16_k(const float* __restrict__ in, unsigned int* __restrict__ out,
                           long long npairs) {
  long long i = (long long)blockIdx.x * blockDim.x + threadIdx.x;
  if (i >= npairs) return;
  unsigned lo = f32_to_bf16_rne(in[2 * i]);
  unsigned hi = f32_to_bf16_rne(in[2 * i + 1]);
  out[i] = lo | (hi << 16);
}

// ---------------- pre-transpose W[l][k][j] -> Wt[l][j][k] bf16 ----------------
__global__ void prep_wt_k(const float* __restrict__ Ws, unsigned short* __restrict__ Wt) {
  int gid = blockIdx.x * blockDim.x + threadIdx.x;
  if (gid >= LAYERS * HID * HID) return;
  int l = gid / (HID * HID);
  int rem = gid - l * HID * HID;
  int j = rem >> 7, k = rem & 127;
  Wt[gid] = f32_to_bf16_rne(Ws[l * HID * HID + k * HID + j]);
}

// ---------------- GEMM: out[n][j] = sum_k Abf[n][k]*Wt[j][k] + bias[j] -----
// One wave per (64-row block, 16-col tile). B fragments for all 4 K-steps are
// loaded once and stay in registers; 16 WMMA issues per wave. Buffers are
// padded to 64 rows so stores need no guards (straight-line b32 store clause).
__global__ void gemm_wmma_k(const unsigned short* __restrict__ Abf,
                            const unsigned short* __restrict__ Wt,
                            const float* __restrict__ bias,
                            float* __restrict__ out, int tiles) {
  int waveId = blockIdx.x * 4 + (threadIdx.x >> 5);
  if (waveId >= tiles) return;               // wave-uniform: EXEC stays all-ones
  int lane = threadIdx.x & 31;
  int rowBlk  = waveId >> 3;                 // HID/16 = 8 column tiles
  int colTile = waveId & 7;
  int m  = lane & 15;
  int hi = lane >> 4;
  int col = colTile * 16 + m;                // B/D column for this lane

  // B 32x16 bf16 fragments, all K-steps: lane holds col; elts 0-15 -> K = k0+16*hi+i
  union { uint4 u[2]; v16bf v; } b[4];
  const unsigned short* brp = Wt + (size_t)col * HID;
  #pragma unroll
  for (int ks = 0; ks < 4; ++ks) {
    b[ks].u[0] = *(const uint4*)(brp + ks * 32 + 16 * hi);
    b[ks].u[1] = *(const uint4*)(brp + ks * 32 + 16 * hi + 8);
  }
  float bb = bias[col];

  #pragma unroll
  for (int rt = 0; rt < MT; ++rt) {
    int row0 = (rowBlk * MT + rt) * 16;
    const unsigned short* arp = Abf + (size_t)(row0 + m) * HID;
    v8f c = {};
    #pragma unroll
    for (int ks = 0; ks < 4; ++ks) {
      // A 16x32 bf16: elts 0-15 -> K = k0+16*hi+i
      union { uint4 u[2]; v16bf v; } a;
      a.u[0] = *(const uint4*)(arp + ks * 32 + 16 * hi);
      a.u[1] = *(const uint4*)(arp + ks * 32 + 16 * hi + 8);
      c = __builtin_amdgcn_wmma_f32_16x16x32_bf16(false, a.v, false, b[ks].v,
                                                  (short)0, c, false, false);
    }
    // D: lane holds col; row = 8*hi + r. Unguarded: out is padded to 64 rows.
    float* op = out + (size_t)(row0 + 8 * hi) * HID + col;
    #pragma unroll
    for (int r = 0; r < 8; ++r) op[(size_t)r * HID] = c[r] + bb;
  }
}

// ---------------- GraphNorm ----------------
__global__ void gn_sum_k(const float* __restrict__ h, const int* __restrict__ batch,
                         float* __restrict__ meanb, float* __restrict__ cntb, int N_) {
  long long gid = (long long)blockIdx.x * blockDim.x + threadIdx.x;
  if (gid >= (long long)N_ * HID) return;
  int n = (int)(gid >> 7), j = (int)(gid & 127);
  int g = batch[n];
  unsafeAtomicAdd(meanb + g * HID + j, h[gid]);
  if (j == 0) unsafeAtomicAdd(cntb + g, 1.0f);
}

__global__ void gn_mean_fin_k(float* __restrict__ meanb, const float* __restrict__ cntb) {
  int gid = blockIdx.x * blockDim.x + threadIdx.x;
  if (gid >= GNUM * HID) return;
  float c = fmaxf(cntb[gid >> 7], 1.0f);
  meanb[gid] = meanb[gid] / c;
}

__global__ void gn_center_k(float* __restrict__ h, const int* __restrict__ batch,
                            const float* __restrict__ meanb, float* __restrict__ varb,
                            const float* __restrict__ alpha_p, int N_) {
  long long gid = (long long)blockIdx.x * blockDim.x + threadIdx.x;
  if (gid >= (long long)N_ * HID) return;
  int n = (int)(gid >> 7), j = (int)(gid & 127);
  int g = batch[n];
  float v = h[gid] - alpha_p[0] * meanb[g * HID + j];
  h[gid] = v;
  unsafeAtomicAdd(varb + g * HID + j, v * v);
}

__global__ void gn_var_fin_k(float* __restrict__ varb, const float* __restrict__ cntb) {
  int gid = blockIdx.x * blockDim.x + threadIdx.x;
  if (gid >= GNUM * HID) return;
  float c = fmaxf(cntb[gid >> 7], 1.0f);
  varb[gid] = rsqrtf(varb[gid] / c + EPSV);    // store inv-std
}

__global__ void gn_apply_k(const float* __restrict__ h, const int* __restrict__ batch,
                           const float* __restrict__ istd, const float* __restrict__ gamma,
                           const float* __restrict__ beta, float* __restrict__ out,
                           int N_, int relu) {
  long long gid = (long long)blockIdx.x * blockDim.x + threadIdx.x;
  if (gid >= (long long)N_ * HID) return;
  int n = (int)(gid >> 7), j = (int)(gid & 127);
  int g = batch[n];
  float v = h[gid] * istd[g * HID + j] * gamma[j] + beta[j];
  if (relu) v = fmaxf(v, 0.0f);
  out[gid] = v;
}

// ---------------------------------------------------------------------------
extern "C" void kernel_launch(void* const* d_in, const int* in_sizes, int n_in,
                              void* d_out, int out_size, void* d_ws, size_t ws_size,
                              hipStream_t stream) {
  const float* x      = (const float*)d_in[0];
  const int*   er     = (const int*)d_in[1];
  const int*   ec     = (const int*)d_in[2];
  const float* ev     = (const float*)d_in[3];
  const int*   batch  = (const int*)d_in[4];
  const float* W_in   = (const float*)d_in[5];
  const float* b_in   = (const float*)d_in[6];
  const float* Ws     = (const float*)d_in[7];
  const float* bs     = (const float*)d_in[8];
  const float* alphas = (const float*)d_in[9];
  const float* gammas = (const float*)d_in[10];
  const float* betas  = (const float*)d_in[11];
  float* out = (float*)d_out;

  const int N = in_sizes[4];          // batch array length = num nodes
  const int E = in_sizes[1];          // edge_row length
  const int N64 = (N + 63) & ~63;     // pad to 64 rows (MT*16) -> unguarded GEMM stores

  // carve workspace (256B aligned)
  char* ws = (char*)d_ws;
  size_t off = 0;
  auto take = [&](size_t bytes) -> char* {
    char* p = ws + off;
    off = (off + bytes + 255) & ~(size_t)255;
    return p;
  };
  float*          H    = (float*)take((size_t)N64 * HID * 4);
  float*          AGG  = (float*)take((size_t)N64 * HID * 4);   // also GEMM out + centered h
  unsigned short* ABF  = (unsigned short*)take((size_t)N64 * HID * 2);
  unsigned short* Wt   = (unsigned short*)take((size_t)LAYERS * HID * HID * 2);
  float*          stat = (float*)take((size_t)(2 * GNUM * HID + GNUM) * 4);
  float* meanb = stat;
  float* varb  = stat + GNUM * HID;
  float* cntb  = stat + 2 * GNUM * HID;

  const long long NH = (long long)N * HID;
  const int TB = 256;
  const int nhBlocks   = (int)((NH + TB - 1) / TB);
  const int pairBlocks = (int)((NH / 2 + TB - 1) / TB);
  const int ghBlocks   = (GNUM * HID + TB - 1) / TB;
  const int tiles      = (N64 / (16 * MT)) * (HID / 16);   // 64-row blocks x 8 col tiles
  const int gemmBlocks = (tiles + 3) / 4;                  // 4 waves (128 threads) per block

  // weights bf16 transpose (deterministic each call)
  prep_wt_k<<<(LAYERS * HID * HID + TB - 1) / TB, TB, 0, stream>>>(Ws, Wt);

  // embed + input projection -> H
  embed_k<<<(N + NPB - 1) / NPB, HID, 0, stream>>>(x, W_in, b_in, H, N);

  for (int l = 0; l < LAYERS; ++l) {
    // SpMM: AGG = A_sparse @ H
    zero_k<<<nhBlocks, TB, 0, stream>>>(AGG, NH);
    spmm_k<<<(int)(((long long)E * 32 + TB - 1) / TB), TB, 0, stream>>>(er, ec, ev, H, AGG, E);

    // bf16 staging + WMMA GEMM (writes back over AGG; padded rows harmless)
    cvt_bf16_k<<<pairBlocks, TB, 0, stream>>>(AGG, (unsigned int*)ABF, NH / 2);
    gemm_wmma_k<<<gemmBlocks, 128, 0, stream>>>(ABF, Wt + (size_t)l * HID * HID,
                                                bs + l * HID, AGG, tiles);

    // GraphNorm
    zero_k<<<(2 * GNUM * HID + GNUM + TB - 1) / TB, TB, 0, stream>>>(stat,
                                                                     2 * GNUM * HID + GNUM);
    gn_sum_k<<<nhBlocks, TB, 0, stream>>>(AGG, batch, meanb, cntb, N);
    gn_mean_fin_k<<<ghBlocks, TB, 0, stream>>>(meanb, cntb);
    gn_center_k<<<nhBlocks, TB, 0, stream>>>(AGG, batch, meanb, varb, alphas + l, N);
    gn_var_fin_k<<<ghBlocks, TB, 0, stream>>>(varb, cntb);

    float* dst = (l == LAYERS - 1) ? out : H;
    gn_apply_k<<<nhBlocks, TB, 0, stream>>>(AGG, batch, varb, gammas + l * HID,
                                            betas + l * HID, dst, N, l < LAYERS - 1 ? 1 : 0);
  }
  (void)n_in; (void)out_size; (void)ws_size;
}
